// HPB_15942918602877
// MI455X (gfx1250) — matrix operand order repaired and checked by reading, not compiled, not run on hardware
//
#include <hip/hip_runtime.h>
#include <hip/hip_bf16.h>
#include <cstdint>

// ---------------- dims ----------------
#define PTOT   32768      // 8 * 64 * 64 positions
#define NB     8
#define DIMC   256
#define NHEADS 8
#define DH     32
#define FFIC   1024
#define HW     4096
#define HH     64
#define WW     64

typedef __attribute__((ext_vector_type(16))) _Float16 v16h;
typedef __attribute__((ext_vector_type(8)))  _Float16 v8h;
typedef __attribute__((ext_vector_type(8)))  float    v8f;

// ---------------- WMMA fragment helpers (CDNA5 16x16x32 f16 layouts) ----------------
// A matrix 16x32 f16, row-major [M x K] source. Per ISA: lanes 0-15 rows M=0..15
// hold K {0..7,16..23}; lanes 16-31 hold K {8..15,24..31}.
__device__ __forceinline__ v16h load_a_frag(const _Float16* A, int lda, int m0, int k0, int lane) {
  int row = m0 + (lane & 15);
  const _Float16* p = A + (size_t)row * lda + k0 + ((lane & 16) ? 8 : 0);
  v8h lo = *(const v8h*)p;
  v8h hi = *(const v8h*)(p + 16);
  v16h r;
#pragma unroll
  for (int i = 0; i < 8; ++i) { r[i] = lo[i]; r[i + 8] = hi[i]; }
  return r;
}

// B matrix 32x16 (K x N) from Bt stored [N x K] row-major. Per ISA: lane n=lane&15,
// lanes 0-15 hold K {0..7,16..23}; lanes 16-31 hold K {8..15,24..31}.
__device__ __forceinline__ v16h load_b_frag(const _Float16* Bt, int ldb, int n0, int k0, int lane) {
  int col = n0 + (lane & 15);
  const _Float16* p = Bt + (size_t)col * ldb + k0 + ((lane & 16) ? 8 : 0);
  v8h lo = *(const v8h*)p;
  v8h hi = *(const v8h*)(p + 16);
  v16h r;
#pragma unroll
  for (int i = 0; i < 8; ++i) { r[i] = lo[i]; r[i + 8] = hi[i]; }
  return r;
}

__device__ __forceinline__ v8f wmma16(v16h a, v16h b, v8f c) {
  return __builtin_amdgcn_wmma_f32_16x16x32_f16(false, a, false, b, (short)0, c, false, false);
}

// ---------------- generic WMMA GEMM: C[M x N] = A[M x K] * Bt[N x K]^T ----------------
// M = PTOT fixed. Each wave owns a 16-row stripe; 4 n-tiles (64 cols) share each
// A fragment (4x less A traffic). Epilogue: optional bias, NCHW residual, f32/f16 store.
template <int N, int K, bool BIAS, bool RES, bool W32, bool W16>
__global__ __launch_bounds__(256) void k_gemm(const _Float16* __restrict__ A,
                                              const _Float16* __restrict__ Bt,
                                              const float* __restrict__ bias,
                                              const float* __restrict__ res,
                                              float* __restrict__ C32,
                                              _Float16* __restrict__ C16,
                                              int ldc) {
  const int lane = threadIdx.x & 31;
  const int wave = threadIdx.x >> 5;
  const int m0 = (blockIdx.x * 8 + wave) * 16;
  const int nlo = lane & 15;
  const int mb = m0 + ((lane & 16) ? 8 : 0);

  for (int n0 = 0; n0 < N; n0 += 64) {
    v8f acc[4];
#pragma unroll
    for (int t = 0; t < 4; ++t) acc[t] = (v8f){};
#pragma unroll 4
    for (int k0 = 0; k0 < K; k0 += 32) {
      if (k0 + 32 < K)
        __builtin_prefetch(A + (size_t)(m0 + nlo) * K + k0 + 32, 0, 1);
      v16h a = load_a_frag(A, K, m0, k0, lane);
#pragma unroll
      for (int t = 0; t < 4; ++t) {
        v16h b = load_b_frag(Bt, K, n0 + t * 16, k0, lane);
        acc[t] = wmma16(a, b, acc[t]);
      }
    }
#pragma unroll
    for (int t = 0; t < 4; ++t) {
      const int n = n0 + t * 16 + nlo;
      const float bv = BIAS ? bias[n] : 0.f;
#pragma unroll
      for (int r = 0; r < 8; ++r) {
        const int m = mb + r;
        float v = acc[t][r] + bv;
        if (RES) {
          const int bb = m >> 12, hw = m & 4095;
          v += res[((size_t)(bb * N + n)) * HW + hw];
        }
        if (W32) C32[(size_t)m * ldc + n] = v;
        if (W16) C16[(size_t)m * ldc + n] = (_Float16)v;
      }
    }
  }
}

// ---------------- helpers ----------------
__global__ void k_f32_to_f16(const float* __restrict__ in, _Float16* __restrict__ out, int n) {
  int i = blockIdx.x * 256 + threadIdx.x;
  if (i < n) out[i] = (_Float16)in[i];
}

// channel layernorm stats: mean/rstd per position
__global__ __launch_bounds__(256) void k_ln_stats(const float* __restrict__ x,
                                                  float* __restrict__ mean,
                                                  float* __restrict__ rstd) {
  int p = blockIdx.x * 256 + threadIdx.x;
  int b = p >> 12, hw = p & 4095;
  const float* px = x + (size_t)b * DIMC * HW + hw;
  float s = 0.f, q = 0.f;
  for (int c = 0; c < DIMC; ++c) {
    float v = px[(size_t)c * HW];
    s += v; q += v * v;
  }
  float m = s * (1.f / DIMC);
  float var = q * (1.f / DIMC) - m * m;
  mean[p] = m;
  rstd[p] = rsqrtf(var + 1e-5f);
}

// apply LN + transpose NCHW -> position-major f16 [P x 256]
__global__ __launch_bounds__(1024) void k_ln_apply(const float* __restrict__ x,
                                                   const float* __restrict__ mean,
                                                   const float* __restrict__ rstd,
                                                   const float* __restrict__ g,
                                                   const float* __restrict__ beta,
                                                   _Float16* __restrict__ xn) {
  __shared__ float tile[32][33];
  int p0 = blockIdx.x * 32;
  int c0 = blockIdx.y * 32;
  int tx = threadIdx.x, ty = threadIdx.y;
  int p = p0 + tx, c = c0 + ty;
  int b = p >> 12, hw = p & 4095;
  float v = x[((size_t)(b * DIMC + c)) * HW + hw];
  tile[ty][tx] = (v - mean[p]) * rstd[p] * g[c] + beta[c];
  __syncthreads();
  p = p0 + ty; c = c0 + tx;
  xn[(size_t)p * DIMC + c] = (_Float16)tile[tx][ty];
}

// per-(bh,d,h) row l2-norm of q,k over W, and scaled row sums
__global__ __launch_bounds__(256) void k_qk_rownorm(const float* __restrict__ qkv,
                                                    float* __restrict__ qscale,
                                                    float* __restrict__ kscale,
                                                    float* __restrict__ qrow,
                                                    float* __restrict__ krow) {
  int r = blockIdx.x * 256 + threadIdx.x;     // 64*32*64 = 131072
  int bh = r >> 11, d = (r >> 6) & 31, h = r & 63;
  int b = bh >> 3, head = bh & 7;
  const float* base = qkv + ((size_t)(b * HW + h * WW)) * 768 + head * DH + d;
  float qs = 0.f, qss = 0.f, ks = 0.f, kss = 0.f;
  for (int w = 0; w < WW; ++w) {
    float q = base[(size_t)w * 768];
    float k = base[(size_t)w * 768 + 256];
    qs += q; qss += q * q;
    ks += k; kss += k * k;
  }
  float qsc = 1.f / fmaxf(sqrtf(qss), 1e-12f);
  float ksc = 1.f / fmaxf(sqrtf(kss), 1e-12f);
  qscale[r] = qsc; kscale[r] = ksc;
  qrow[r] = qs * qsc; krow[r] = ks * ksc;
}

// q_probe and row scores per bh
__global__ __launch_bounds__(64) void k_score_r(const float* __restrict__ qrow,
                                                const float* __restrict__ krow,
                                                float* __restrict__ qprobe,
                                                float* __restrict__ score_r) {
  __shared__ float qp[32];
  int bh = blockIdx.x, t = threadIdx.x;
  if (t < 32) {
    float s = 0.f;
    const float* q = qrow + ((size_t)bh * 32 + t) * 64;
    for (int h = 0; h < 64; ++h) s += q[h];
    qp[t] = s;
    qprobe[bh * 32 + t] = s;
  }
  __syncthreads();
  float sc = 0.f;
  const float* kr = krow + (size_t)bh * 32 * 64;
  for (int d = 0; d < 32; ++d) sc += qp[d] * kr[d * 64 + t];
  score_r[bh * 64 + t] = sc;
}

// top-16 of 64 (descending set selection; ordering irrelevant downstream)
__global__ void k_topk(const float* __restrict__ score, int* __restrict__ idx) {
  int bh = blockIdx.x;
  const float* s = score + bh * 64;
  int* o = idx + bh * 16;
  unsigned long long used = 0ull;
  for (int t = 0; t < 16; ++t) {
    float best = -3.0e38f; int bi = 0;
    for (int j = 0; j < 64; ++j)
      if (!((used >> j) & 1ull) && s[j] > best) { best = s[j]; bi = j; }
    used |= 1ull << bi;
    o[t] = bi;
  }
}

// column scores after row pruning
__global__ __launch_bounds__(64) void k_score_c(const float* __restrict__ qkv,
                                                const float* __restrict__ kscale,
                                                const float* __restrict__ qprobe,
                                                const int* __restrict__ idx_h,
                                                float* __restrict__ score_c) {
  int bh = blockIdx.x, w = threadIdx.x;
  int b = bh >> 3, head = bh & 7;
  float sc = 0.f;
  for (int d = 0; d < 32; ++d) {
    float kw = 0.f;
    for (int hi = 0; hi < 16; ++hi) {
      int h = idx_h[bh * 16 + hi];
      float kraw = qkv[((size_t)(b * HW + h * WW + w)) * 768 + 256 + head * DH + d];
      kw += kraw * kscale[((size_t)bh * 32 + d) * 64 + h];
    }
    sc += qprobe[bh * 32 + d] * kw;
  }
  score_c[bh * 64 + w] = sc;
}

// qf_h[bh][i][32] f16 (normalized q, position-major per bh)
__global__ void k_build_qf(const float* __restrict__ qkv,
                           const float* __restrict__ qscale,
                           _Float16* __restrict__ qf) {
  size_t id = (size_t)blockIdx.x * 256 + threadIdx.x;   // 64*4096*32
  int d = id & 31;
  size_t t = id >> 5;
  int i = t & 4095;
  int bh = (int)(t >> 12);
  int b = bh >> 3, head = bh & 7, h = i >> 6;
  float q = qkv[((size_t)(b * HW + i)) * 768 + head * DH + d];
  qf[id] = (_Float16)(q * qscale[((size_t)bh * 32 + d) * 64 + h]);
}

// pruned kf [bh][256][32] and vfT [bh][32][256] f16
__global__ void k_build_kv(const float* __restrict__ qkv,
                           const float* __restrict__ kscale,
                           const int* __restrict__ idx_h,
                           const int* __restrict__ idx_w,
                           _Float16* __restrict__ kf,
                           _Float16* __restrict__ vfT) {
  int id = blockIdx.x * 256 + threadIdx.x;    // 64*256*32
  int d = id & 31;
  int t = id >> 5;
  int j = t & 255;
  int bh = t >> 8;
  int hi = j >> 4, wi = j & 15;
  int h = idx_h[bh * 16 + hi], w = idx_w[bh * 16 + wi];
  int b = bh >> 3, head = bh & 7;
  size_t p = (size_t)(b * HW + h * WW + w);
  float kraw = qkv[p * 768 + 256 + head * DH + d];
  float vraw = qkv[p * 768 + 512 + head * DH + d];
  kf[((size_t)bh * 256 + j) * 32 + d] = (_Float16)(kraw * kscale[((size_t)bh * 32 + d) * 64 + h]);
  vfT[((size_t)bh * 32 + d) * 256 + j] = (_Float16)vraw;
}

// attention core: per bh, 256 keys staged into LDS via gfx1250 async memory->LDS
// copies (ASYNCcnt), sim held in accumulation VGPRs, wave32 shuffle softmax,
// attn probs through LDS back into WMMA A-fragments.
__global__ __launch_bounds__(128) void k_attn(const _Float16* __restrict__ qf,
                                              const _Float16* __restrict__ kf,
                                              const _Float16* __restrict__ vfT,
                                              _Float16* __restrict__ attnA) {
  __shared__ _Float16 smem_kf[256 * 32];
  __shared__ _Float16 smem_vt[32 * 256];
  __shared__ _Float16 smem_at[4 * 16 * 256];

  const int bh = blockIdx.y;
  const int lane = threadIdx.x & 31;
  const int wave = threadIdx.x >> 5;

  // --- async memory->LDS staging (GLOBAL_LOAD_ASYNC_TO_LDS_B128, ASYNCcnt) ---
  {
    const _Float16* kg = kf + (size_t)bh * 256 * 32;
    const _Float16* vg = vfT + (size_t)bh * 32 * 256;
    unsigned kf_lds = (unsigned)(uintptr_t)(void*)smem_kf;   // LDS byte offset
    unsigned vt_lds = (unsigned)(uintptr_t)(void*)smem_vt;
#pragma unroll
    for (int it = 0; it < 8; ++it) {
      int e = it * 1024 + threadIdx.x * 8;                   // halves, 16B per lane
      asm volatile("global_load_async_to_lds_b128 %0, %1, off"
                   :: "v"(kf_lds + e * 2), "v"(kg + e) : "memory");
      asm volatile("global_load_async_to_lds_b128 %0, %1, off"
                   :: "v"(vt_lds + e * 2), "v"(vg + e) : "memory");
    }
    asm volatile("s_wait_asynccnt 0" ::: "memory");
  }
  __syncthreads();

  const int i0 = (blockIdx.x * 4 + wave) * 16;
  const _Float16* Aq = qf + ((size_t)bh * 4096 + i0) * 32;
  v16h afrag = load_a_frag(Aq, 32, 0, 0, lane);

  // sim = q x k^T : 16 j-tiles, K=32 (one WMMA each), A fragment reused.
  v8f acc[16];
#pragma unroll
  for (int jt = 0; jt < 16; ++jt) acc[jt] = (v8f){};
#pragma unroll
  for (int jt = 0; jt < 16; ++jt) {
    v16h bfrag = load_b_frag(smem_kf, 32, jt * 16, 0, lane);
    acc[jt] = wmma16(afrag, bfrag, acc[jt]);
  }

  // softmax over 256 keys: row m lives in vgpr r across 16 lanes of the half-wave.
  float invsum[8];
#pragma unroll
  for (int r = 0; r < 8; ++r) {
    float m = -3.0e38f;
#pragma unroll
    for (int jt = 0; jt < 16; ++jt) m = fmaxf(m, acc[jt][r]);
    m = fmaxf(m, __shfl_xor(m, 1, 32));
    m = fmaxf(m, __shfl_xor(m, 2, 32));
    m = fmaxf(m, __shfl_xor(m, 4, 32));
    m = fmaxf(m, __shfl_xor(m, 8, 32));
    float s = 0.f;
#pragma unroll
    for (int jt = 0; jt < 16; ++jt) {
      float e = __expf(acc[jt][r] - m);
      acc[jt][r] = e;
      s += e;
    }
    s += __shfl_xor(s, 1, 32);
    s += __shfl_xor(s, 2, 32);
    s += __shfl_xor(s, 4, 32);
    s += __shfl_xor(s, 8, 32);
    invsum[r] = 1.f / s;
  }

  const int nlo = lane & 15;
  const int mbase = (lane & 16) ? 8 : 0;
  _Float16* attw = smem_at + (size_t)wave * 16 * 256;
#pragma unroll
  for (int jt = 0; jt < 16; ++jt)
#pragma unroll
    for (int r = 0; r < 8; ++r)
      attw[(mbase + r) * 256 + jt * 16 + nlo] = (_Float16)(acc[jt][r] * invsum[r]);

  // out = attn x V : attn [16 x 256] as A, vfT [32 x 256] as Bt, K stepped by 32.
  const int b = bh >> 3, head = bh & 7;
#pragma unroll
  for (int nt = 0; nt < 2; ++nt) {
    v8f o = {};
#pragma unroll
    for (int kt = 0; kt < 8; ++kt) {
      v16h a = load_a_frag(attw, 256, 0, kt * 32, lane);
      v16h bfr = load_b_frag(smem_vt, 256, nt * 16, kt * 32, lane);
      o = wmma16(a, bfr, o);
    }
    const int d = nt * 16 + nlo;
#pragma unroll
    for (int r = 0; r < 8; ++r) {
      const int i = i0 + mbase + r;
      attnA[((size_t)(b * HW + i)) * 256 + head * DH + d] = (_Float16)o[r];
    }
  }
}

// depthwise 3x3 on x (NCHW), writes f16 into combined[:, 256:512]
__global__ __launch_bounds__(256) void k_dw_x(const float* __restrict__ x,
                                              const float* __restrict__ wdw,
                                              const float* __restrict__ bdw,
                                              _Float16* __restrict__ combined) {
  size_t id = (size_t)blockIdx.x * 256 + threadIdx.x;  // 8*256*4096
  int hw = id & 4095;
  int c = (int)((id >> 12) & 255);
  int b = (int)(id >> 20);
  int h = hw >> 6, w = hw & 63;
  const float* xin = x + ((size_t)(b * DIMC + c)) * HW;
  const float* wk = wdw + c * 9;
  float acc = bdw[c];
#pragma unroll
  for (int dh = -1; dh <= 1; ++dh)
#pragma unroll
    for (int dw = -1; dw <= 1; ++dw) {
      int hh = h + dh, ww2 = w + dw;
      if (hh >= 0 && hh < HH && ww2 >= 0 && ww2 < WW)
        acc += xin[hh * WW + ww2] * wk[(dh + 1) * 3 + (dw + 1)];
    }
  combined[((size_t)(b * HW + hw)) * 512 + 256 + c] = (_Float16)acc;
}

// depthwise 3x3 on h (position-major [P x 1024]), f32 out
__global__ __launch_bounds__(256) void k_dw_h(const float* __restrict__ hbuf,
                                              const float* __restrict__ wdw,
                                              const float* __restrict__ bdw,
                                              float* __restrict__ rpre) {
  size_t id = (size_t)blockIdx.x * 256 + threadIdx.x;  // 32768*1024
  int c = id & 1023;
  size_t p = id >> 10;
  int hw = (int)(p & 4095);
  size_t pb = p - hw;
  int h = hw >> 6, w = hw & 63;
  const float* wk = wdw + c * 9;
  float acc = bdw[c];
#pragma unroll
  for (int dh = -1; dh <= 1; ++dh)
#pragma unroll
    for (int dw = -1; dw <= 1; ++dw) {
      int hh = h + dh, ww2 = w + dw;
      if (hh >= 0 && hh < HH && ww2 >= 0 && ww2 < WW)
        acc += hbuf[(pb + hh * WW + ww2) * FFIC + c] * wk[(dh + 1) * 3 + (dw + 1)];
    }
  rpre[id] = acc;
}

// instance-norm stats per (b, c) over 4096 positions, position-major input
template <int CC>
__global__ __launch_bounds__(256) void k_in_stats(const float* __restrict__ in,
                                                  float* __restrict__ mean,
                                                  float* __restrict__ rstd) {
  int bc = blockIdx.x;
  int b = bc / CC, c = bc % CC;
  __shared__ float s1[256], s2[256];
  float a = 0.f, q = 0.f;
  for (int t = threadIdx.x; t < HW; t += 256) {
    float v = in[((size_t)(b * HW + t)) * CC + c];
    a += v; q += v * v;
  }
  s1[threadIdx.x] = a; s2[threadIdx.x] = q;
  __syncthreads();
  for (int o = 128; o > 0; o >>= 1) {
    if (threadIdx.x < o) {
      s1[threadIdx.x] += s1[threadIdx.x + o];
      s2[threadIdx.x] += s2[threadIdx.x + o];
    }
    __syncthreads();
  }
  if (threadIdx.x == 0) {
    float m = s1[0] * (1.f / HW);
    float var = s2[0] * (1.f / HW) - m * m;
    mean[bc] = m;
    rstd[bc] = rsqrtf(var + 1e-5f);
  }
}

__device__ __forceinline__ float gelu_exact(float v) {
  return 0.5f * v * (1.f + erff(v * 0.70710678118654752f));
}

// h = gelu(instnorm(h_pre))
__global__ void k_gelu_h(const float* __restrict__ hpre,
                         const float* __restrict__ mean,
                         const float* __restrict__ rstd,
                         float* __restrict__ hout) {
  size_t id = (size_t)blockIdx.x * 256 + threadIdx.x;
  int c = id & 1023;
  size_t p = id >> 10;
  int b = (int)(p >> 12);
  int bc = b * FFIC + c;
  float v = (hpre[id] - mean[bc]) * rstd[bc];
  hout[id] = gelu_exact(v);
}

// h2 = h + gelu(instnorm(r_pre)), f16 out for ff2 GEMM
__global__ void k_addgelu(const float* __restrict__ hbuf,
                          const float* __restrict__ rpre,
                          const float* __restrict__ mean,
                          const float* __restrict__ rstd,
                          _Float16* __restrict__ h2) {
  size_t id = (size_t)blockIdx.x * 256 + threadIdx.x;
  int c = id & 1023;
  size_t p = id >> 10;
  int b = (int)(p >> 12);
  int bc = b * FFIC + c;
  float v = (rpre[id] - mean[bc]) * rstd[bc];
  h2[id] = (_Float16)(hbuf[id] + gelu_exact(v));
}

// final instnorm, write NCHW f32
__global__ void k_final(const float* __restrict__ ypre,
                        const float* __restrict__ mean,
                        const float* __restrict__ rstd,
                        float* __restrict__ out) {
  size_t id = (size_t)blockIdx.x * 256 + threadIdx.x;  // 8*256*4096, hw fastest
  int hw = id & 4095;
  int c = (int)((id >> 12) & 255);
  int b = (int)(id >> 20);
  float v = ypre[((size_t)(b * HW + hw)) * DIMC + c];
  int bc = b * DIMC + c;
  out[id] = (v - mean[bc]) * rstd[bc];
}

// ---------------- host launch ----------------
extern "C" void kernel_launch(void* const* d_in, const int* in_sizes, int n_in,
                              void* d_out, int out_size, void* d_ws, size_t ws_size,
                              hipStream_t stream) {
  (void)in_sizes; (void)n_in; (void)out_size; (void)ws_size;
  const float* x      = (const float*)d_in[0];
  const float* ln_g   = (const float*)d_in[1];
  const float* ln_b   = (const float*)d_in[2];
  const float* w_qkv  = (const float*)d_in[3];
  const float* w_out  = (const float*)d_in[4];
  const float* b_out  = (const float*)d_in[5];
  const float* w_dw   = (const float*)d_in[6];
  const float* b_dw   = (const float*)d_in[7];
  const float* w_comb = (const float*)d_in[8];
  const float* b_comb = (const float*)d_in[9];
  const float* w_ff1  = (const float*)d_in[10];
  const float* b_ff1  = (const float*)d_in[11];
  const float* w_ffdw = (const float*)d_in[12];
  const float* b_ffdw = (const float*)d_in[13];
  const float* w_ff2  = (const float*)d_in[14];
  const float* b_ff2  = (const float*)d_in[15];
  float* out = (float*)d_out;

  uintptr_t base = (uintptr_t)d_ws;
  size_t off = 0;
  auto alloc = [&](size_t bytes) -> void* {
    void* p = (void*)(base + off);
    off += (bytes + 255) & ~(size_t)255;
    return p;
  };

  float* ln_mean = (float*)alloc((size_t)PTOT * 4);
  float* ln_rstd = (float*)alloc((size_t)PTOT * 4);
  _Float16* wqkv_h = (_Float16*)alloc((size_t)768 * 256 * 2);
  _Float16* wout_h = (_Float16*)alloc((size_t)256 * 256 * 2);
  _Float16* wcomb_h = (_Float16*)alloc((size_t)256 * 512 * 2);
  _Float16* wff1_h = (_Float16*)alloc((size_t)1024 * 256 * 2);
  _Float16* wff2_h = (_Float16*)alloc((size_t)256 * 1024 * 2);
  // region1: qkv (P x 768 f32) then h_pre (P x 1024 f32)
  void* region1 = alloc((size_t)PTOT * 1024 * 4);
  float* qkv = (float*)region1;
  float* h_pre = (float*)region1;
  // region2: combined (P x 512 f16) then y_pre (P x 256 f32) -- same byte size
  void* region2 = alloc((size_t)PTOT * 512 * 2);
  _Float16* combined = (_Float16*)region2;
  float* y_pre = (float*)region2;
  // region3: xn_h (P x 256 f16) then attn_out_h (P x 256 f16)
  void* region3 = alloc((size_t)PTOT * 256 * 2);
  _Float16* xn_h = (_Float16*)region3;
  _Float16* aout_h = (_Float16*)region3;

  float* qscale = (float*)alloc((size_t)131072 * 4);
  float* kscale = (float*)alloc((size_t)131072 * 4);
  float* qrow   = (float*)alloc((size_t)131072 * 4);
  float* krow   = (float*)alloc((size_t)131072 * 4);
  float* qprobe = (float*)alloc((size_t)64 * 32 * 4);
  float* score_r = (float*)alloc((size_t)64 * 64 * 4);
  float* score_c = (float*)alloc((size_t)64 * 64 * 4);
  int* idx_h = (int*)alloc((size_t)64 * 16 * 4);
  int* idx_w = (int*)alloc((size_t)64 * 16 * 4);
  _Float16* qf = (_Float16*)alloc((size_t)64 * 4096 * 32 * 2);
  _Float16* kf = (_Float16*)alloc((size_t)64 * 256 * 32 * 2);
  _Float16* vfT = (_Float16*)alloc((size_t)64 * 32 * 256 * 2);
  _Float16* attnA = (_Float16*)alloc((size_t)PTOT * 256 * 2);
  float* hbuf = (float*)alloc((size_t)PTOT * 1024 * 4);
  float* r_pre = (float*)alloc((size_t)PTOT * 1024 * 4);
  _Float16* h2_h = (_Float16*)alloc((size_t)PTOT * 1024 * 2);
  float* ist_m = (float*)alloc((size_t)8 * 1024 * 4);
  float* ist_r = (float*)alloc((size_t)8 * 1024 * 4);

  // 1. weight conversion f32 -> f16
  k_f32_to_f16<<<(768 * 256 + 255) / 256, 256, 0, stream>>>(w_qkv, wqkv_h, 768 * 256);
  k_f32_to_f16<<<(256 * 256 + 255) / 256, 256, 0, stream>>>(w_out, wout_h, 256 * 256);
  k_f32_to_f16<<<(256 * 512 + 255) / 256, 256, 0, stream>>>(w_comb, wcomb_h, 256 * 512);
  k_f32_to_f16<<<(1024 * 256 + 255) / 256, 256, 0, stream>>>(w_ff1, wff1_h, 1024 * 256);
  k_f32_to_f16<<<(256 * 1024 + 255) / 256, 256, 0, stream>>>(w_ff2, wff2_h, 256 * 1024);

  // 2. channel layernorm -> xn_h [P x 256] f16
  k_ln_stats<<<PTOT / 256, 256, 0, stream>>>(x, ln_mean, ln_rstd);
  k_ln_apply<<<dim3(PTOT / 32, DIMC / 32), dim3(32, 32), 0, stream>>>(x, ln_mean, ln_rstd, ln_g, ln_b, xn_h);

  // 3. qkv GEMM [P x 256] x [256 x 768]
  k_gemm<768, 256, false, false, true, false><<<PTOT / 128, 256, 0, stream>>>(
      xn_h, wqkv_h, nullptr, nullptr, qkv, nullptr, 768);

  // 4. l2 norms, probes, top-k pruning
  k_qk_rownorm<<<131072 / 256, 256, 0, stream>>>(qkv, qscale, kscale, qrow, krow);
  k_score_r<<<64, 64, 0, stream>>>(qrow, krow, qprobe, score_r);
  k_topk<<<64, 1, 0, stream>>>(score_r, idx_h);
  k_score_c<<<64, 64, 0, stream>>>(qkv, kscale, qprobe, idx_h, score_c);
  k_topk<<<64, 1, 0, stream>>>(score_c, idx_w);

  // 5. gather q / pruned k,v as f16
  k_build_qf<<<(64 * 4096 * 32) / 256, 256, 0, stream>>>(qkv, qscale, qf);
  k_build_kv<<<(64 * 256 * 32) / 256, 256, 0, stream>>>(qkv, kscale, idx_h, idx_w, kf, vfT);

  // 6. attention core (async LDS staging + WMMA sim + softmax + WMMA out)
  k_attn<<<dim3(4096 / 64, 64), 128, 0, stream>>>(qf, kf, vfT, attnA);

  // 7. out projection -> combined[:, 0:256] f16 (bias)
  k_gemm<256, 256, true, false, false, true><<<PTOT / 128, 256, 0, stream>>>(
      attnA, wout_h, b_out, nullptr, nullptr, combined, 512);

  // 8. depthwise conv branch -> combined[:, 256:512]
  k_dw_x<<<(NB * DIMC * HW) / 256, 256, 0, stream>>>(x, w_dw, b_dw, combined);

  // 9. combine projection + residual x -> attn_out f16 [P x 256]
  k_gemm<256, 512, true, true, false, true><<<PTOT / 128, 256, 0, stream>>>(
      combined, wcomb_h, b_comb, x, nullptr, aout_h, 256);

  // 10. ff1 -> h_pre f32 [P x 1024]
  k_gemm<1024, 256, true, false, true, false><<<PTOT / 128, 256, 0, stream>>>(
      aout_h, wff1_h, b_ff1, nullptr, h_pre, nullptr, 1024);

  // 11. instnorm + gelu -> h
  k_in_stats<1024><<<8 * 1024, 256, 0, stream>>>(h_pre, ist_m, ist_r);
  k_gelu_h<<<(size_t)(PTOT) * 1024 / 256, 256, 0, stream>>>(h_pre, ist_m, ist_r, hbuf);

  // 12. ff depthwise conv + instnorm + gelu, residual -> h2 f16
  k_dw_h<<<(size_t)(PTOT) * 1024 / 256, 256, 0, stream>>>(hbuf, w_ffdw, b_ffdw, r_pre);
  k_in_stats<1024><<<8 * 1024, 256, 0, stream>>>(r_pre, ist_m, ist_r);
  k_addgelu<<<(size_t)(PTOT) * 1024 / 256, 256, 0, stream>>>(hbuf, r_pre, ist_m, ist_r, h2_h);

  // 13. ff2 -> y_pre f32 [P x 256]
  k_gemm<256, 1024, true, false, true, false><<<PTOT / 128, 256, 0, stream>>>(
      h2_h, wff2_h, b_ff2, nullptr, y_pre, nullptr, 256);

  // 14. final instnorm -> NCHW output
  k_in_stats<256><<<8 * 256, 256, 0, stream>>>(y_pre, ist_m, ist_r);
  k_final<<<(NB * DIMC * HW) / 256, 256, 0, stream>>>(y_pre, ist_m, ist_r, out);
}
